// TemporalSelfAttention_26130581028918
// MI455X (gfx1250) — matrix operand order repaired and checked by reading, not compiled, not run on hardware
//
#include <hip/hip_runtime.h>

// ---------------------------------------------------------------------------
// TemporalSelfAttention (BEVFormer-style MS deformable attn) for MI455X
// gfx1250: wave32, WMMA bf16 3x-split fp32 emulation, bf16 staged in LDS.
// ---------------------------------------------------------------------------

typedef unsigned short u16;
typedef __attribute__((ext_vector_type(16))) __bf16 v16bf;
typedef __attribute__((ext_vector_type(2)))  __bf16 v2bf;
typedef __attribute__((ext_vector_type(8)))  float  v8f;

#define HH    200
#define WWID  200
#define NQ    40000
#define DDIM  256
#define NHEAD 8
#define HDIM  32

__device__ __forceinline__ u16 f32_to_bf16_rn(float f) {
  unsigned int u = __float_as_uint(f);
  unsigned int lsb = (u >> 16) & 1u;
  u += 0x7fffu + lsb;               // round-to-nearest-even
  return (u16)(u >> 16);
}
__device__ __forceinline__ float bf16_to_f32(u16 h) {
  return __uint_as_float(((unsigned int)h) << 16);
}

// Split a pair of f32 into packed bf16 hi (RNE) and bf16 lo (residual).
__device__ __forceinline__ void bf16_split_pair(float x0, float x1,
                                                unsigned int& hp, unsigned int& lp) {
#if __has_builtin(__builtin_amdgcn_cvt_pk_bf16_f32)
  v2bf h = __builtin_amdgcn_cvt_pk_bf16_f32(x0, x1);
  unsigned int hu = __builtin_bit_cast(unsigned int, h);
  float h0 = __uint_as_float(hu << 16);
  float h1 = __uint_as_float(hu & 0xffff0000u);
  v2bf l = __builtin_amdgcn_cvt_pk_bf16_f32(x0 - h0, x1 - h1);
  hp = hu;
  lp = __builtin_bit_cast(unsigned int, l);
#else
  unsigned int u0 = __float_as_uint(x0);
  unsigned int u1 = __float_as_uint(x1);
  unsigned int r0 = u0 + 0x7fffu + ((u0 >> 16) & 1u);
  unsigned int r1 = u1 + 0x7fffu + ((u1 >> 16) & 1u);
  hp = (r0 >> 16) | (r1 & 0xffff0000u);
  float l0 = x0 - __uint_as_float(r0 & 0xffff0000u);
  float l1 = x1 - __uint_as_float(r1 & 0xffff0000u);
  unsigned int v0 = __float_as_uint(l0);
  unsigned int v1 = __float_as_uint(l1);
  unsigned int q0 = v0 + 0x7fffu + ((v0 >> 16) & 1u);
  unsigned int q1 = v1 + 0x7fffu + ((v1 >> 16) & 1u);
  lp = (q0 >> 16) | (q1 & 0xffff0000u);
#endif
}

// ---------------------------------------------------------------------------
// Weight prep: W is [K][N] fp32 row-major; emit Wt_hi/Wt_lo as [N][K] bf16
// so GEMM B-fragments are two contiguous 16B loads per lane.
// ---------------------------------------------------------------------------
__global__ void prep_w_kernel(const float* __restrict__ W, u16* __restrict__ hi,
                              u16* __restrict__ lo, int K, int N) {
  int i = blockIdx.x * blockDim.x + threadIdx.x;
  if (i >= K * N) return;
  int n = i / K, k = i - n * K;
  float x = W[(size_t)k * N + n];
  u16 h = f32_to_bf16_rn(x);
  hi[i] = h;
  lo[i] = f32_to_bf16_rn(x - bf16_to_f32(h));
}

// ---------------------------------------------------------------------------
// GEMM: C[M,N] = A[M,K] * B[K,N] + bias (+ addsrc), fp32 in/out.
// A may be split along K at K0 (A rows: k<K0 -> A0, else A1; both stride K0).
// bf16 3x-split WMMA: acc += Ah*Bh + Ah*Bl + Al*Bh  (fp32 accumulate).
// Block = 64x64 tile, 8 waves (2x4), each wave = 32x16 (two 16x16 WMMA tiles).
// A tile is converted to bf16 hi/lo ONCE at staging time; fragment loads are
// pure ds_load_b128 + bit-cast (no per-fragment conversion VALU).
// ---------------------------------------------------------------------------
#define BM 64
#define BN 64
#define LDSA_STRIDE 40   // 32 + 8 pad bf16 (80B row stride -> bank spread)

union AFrag { uint4 q[2]; v16bf v; };

template<int K, int K0, bool ADD_SRC>
__global__ __launch_bounds__(256)
void gemm_bf16x3_kernel(const float* __restrict__ A0, const float* __restrict__ A1,
                        const u16* __restrict__ Bhi, const u16* __restrict__ Blo,
                        const float* __restrict__ bias,
                        const float* __restrict__ addsrc,
                        float* __restrict__ C, int N) {
  __shared__ u16 lds_ahi[BM * LDSA_STRIDE];
  __shared__ u16 lds_alo[BM * LDSA_STRIDE];

  const int tid  = threadIdx.x;
  const int wave = tid >> 5;
  const int lane = tid & 31;
  const int half = lane >> 4;      // WMMA lane-half
  const int l16  = lane & 15;
  const int wm0  = (wave >> 2) * 32;   // wave row base within 64
  const int wn0  = (wave & 3) * 16;    // wave col base within 64
  const int bm0  = blockIdx.x * BM;
  const int bn0  = blockIdx.y * BN;

  // cooperative A-tile staging: 64 rows x 32 f32, 8 floats per thread
  const int srow = tid >> 2;
  const int scol = (tid & 3) * 8;

  v8f acc[2];
  acc[0] = (v8f){0.f,0.f,0.f,0.f,0.f,0.f,0.f,0.f};
  acc[1] = acc[0];

  const int  ncol = bn0 + wn0 + l16;            // this lane's output column
  const u16* bhip = Bhi + (size_t)ncol * K;
  const u16* blop = Blo + (size_t)ncol * K;

  #pragma unroll
  for (int kk = 0; kk < K; kk += 32) {
    // ---- stage A tile: load f32, split to bf16 hi/lo, store packed ----
    const float* Ap = (kk < K0) ? A0 : A1;
    const int    kc = (kk < K0) ? kk : (kk - K0);
    const float* src = Ap + (size_t)(bm0 + srow) * (size_t)K0 + kc + scol;
    float4 s0 = *(const float4*)(src);
    float4 s1 = *(const float4*)(src + 4);
    unsigned int hp[4], lp[4];
    bf16_split_pair(s0.x, s0.y, hp[0], lp[0]);
    bf16_split_pair(s0.z, s0.w, hp[1], lp[1]);
    bf16_split_pair(s1.x, s1.y, hp[2], lp[2]);
    bf16_split_pair(s1.z, s1.w, hp[3], lp[3]);
    *(uint4*)&lds_ahi[srow * LDSA_STRIDE + scol] = make_uint4(hp[0], hp[1], hp[2], hp[3]);
    *(uint4*)&lds_alo[srow * LDSA_STRIDE + scol] = make_uint4(lp[0], lp[1], lp[2], lp[3]);

    if (kk + 32 < K) {  // prefetch next A tile -> global_prefetch_b8
      const float* Apn = ((kk + 32) < K0) ? A0 : A1;
      const int    kcn = ((kk + 32) < K0) ? (kk + 32) : (kk + 32 - K0);
      __builtin_prefetch(Apn + (size_t)(bm0 + srow) * (size_t)K0 + kcn + scol, 0, 1);
    }
    __syncthreads();

    // ---- B fragments: 32x16 bf16, lane-half -> K runs {8h..8h+7} and {16+8h..23+8h} from [N][K] ----
    AFrag bhf, blf;
    bhf.q[0] = *(const uint4*)(bhip + kk + 8 * half);
    bhf.q[1] = *(const uint4*)(bhip + kk + 16 + 8 * half);
    blf.q[0] = *(const uint4*)(blop + kk + 8 * half);
    blf.q[1] = *(const uint4*)(blop + kk + 16 + 8 * half);
    v16bf bh = bhf.v;
    v16bf bl = blf.v;

    #pragma unroll
    for (int mf = 0; mf < 2; ++mf) {
      // A fragment 16x32: lane-half -> K runs {8h..8h+7} and {16+8h..23+8h}
      const int rbase = (wm0 + mf * 16 + l16) * LDSA_STRIDE;
      AFrag ah, al;
      ah.q[0] = *(const uint4*)&lds_ahi[rbase + 8 * half];
      ah.q[1] = *(const uint4*)&lds_ahi[rbase + 16 + 8 * half];
      al.q[0] = *(const uint4*)&lds_alo[rbase + 8 * half];
      al.q[1] = *(const uint4*)&lds_alo[rbase + 16 + 8 * half];
      acc[mf] = __builtin_amdgcn_wmma_f32_16x16x32_bf16(
          false, ah.v, false, bh, (short)0, acc[mf], false, false);
      acc[mf] = __builtin_amdgcn_wmma_f32_16x16x32_bf16(
          false, ah.v, false, bl, (short)0, acc[mf], false, false);
      acc[mf] = __builtin_amdgcn_wmma_f32_16x16x32_bf16(
          false, al.v, false, bh, (short)0, acc[mf], false, false);
    }
    __syncthreads();
  }

  // ---- epilogue: C layout — VGPR r: rows (8*half + r), col = l16 ----
  const float bv = bias[ncol];
  #pragma unroll
  for (int mf = 0; mf < 2; ++mf) {
    #pragma unroll
    for (int r = 0; r < 8; ++r) {
      int row = bm0 + wm0 + mf * 16 + 8 * half + r;
      size_t o = (size_t)row * N + ncol;
      float v = acc[mf][r] + bv;
      if (ADD_SRC) v += addsrc[o];
      C[o] = v;
    }
  }
}

// ---------------------------------------------------------------------------
// Deformable sampling + softmax + bq-mean.
// grid = NQ blocks, 256 threads = 8 waves = 8 heads; lane = channel d (0..31).
// Lanes s<8 each own one (bq,p) sample: softmax over p via shfl_xor within
// groups of 4; gather loop broadcasts (w,x,y) from lane s; all corner loads
// are wave-uniform addresses -> 32-lane coalesced 128B reads (L2-resident).
// ---------------------------------------------------------------------------
__global__ __launch_bounds__(256)
void sample_kernel(const float* __restrict__ vproj,   // [2*NQ][256] (=2,NQ,8,32)
                   const float* __restrict__ off,     // [NQ][128]  (h,bq,p,2)
                   const float* __restrict__ awl,     // [NQ][64]   (h,bq,p)
                   const float* __restrict__ rp,      // [NQ][2]
                   float* __restrict__ outs) {        // [NQ][256]
  const int q    = blockIdx.x;
  const int h    = threadIdx.x >> 5;
  const int lane = threadIdx.x & 31;
  const int s    = lane & 7;                 // replicated sample id (bq*4+p)

  float logit = awl[q * 64 + h * 8 + s];
  float offx  = off[q * 128 + (h * 8 + s) * 2 + 0];
  float offy  = off[q * 128 + (h * 8 + s) * 2 + 1];
  float rx = rp[q * 2 + 0], ry = rp[q * 2 + 1];

  // softmax over the 4 points within each bq group (lanes {0..3},{4..7} pat.)
  float m = logit;
  m = fmaxf(m, __shfl_xor(m, 1, 32));
  m = fmaxf(m, __shfl_xor(m, 2, 32));
  float e = __expf(logit - m);
  float sum = e;
  sum += __shfl_xor(sum, 1, 32);
  sum += __shfl_xor(sum, 2, 32);
  float w = e / sum;

  // loc = (ref + off/norm)*[W,H] - 0.5 = ref*200 + off - 0.5
  float x = rx * (float)WWID + offx - 0.5f;
  float y = ry * (float)HH   + offy - 0.5f;

  float accv = 0.0f;
  #pragma unroll
  for (int si = 0; si < 8; ++si) {
    float ws = __shfl(w, si, 32);
    float xs = __shfl(x, si, 32);
    float ys = __shfl(y, si, 32);
    int bq = si >> 2;
    float x0f = floorf(xs), y0f = floorf(ys);
    float lx = xs - x0f,  ly = ys - y0f;
    int x0 = (int)x0f, y0 = (int)y0f;
    const float* Vb = vproj + (size_t)bq * NQ * DDIM + h * HDIM;
    #pragma unroll
    for (int dy = 0; dy < 2; ++dy) {
      int yi = y0 + dy;
      float wy = dy ? ly : (1.0f - ly);
      #pragma unroll
      for (int dx = 0; dx < 2; ++dx) {
        int xi = x0 + dx;
        float wx = dx ? lx : (1.0f - lx);
        bool valid = (xi >= 0) && (xi < WWID) && (yi >= 0) && (yi < HH);
        int xc = min(max(xi, 0), WWID - 1);
        int yc = min(max(yi, 0), HH - 1);
        float v = Vb[(size_t)(yc * WWID + xc) * DDIM + lane];
        accv += valid ? (ws * wy * wx * v) : 0.0f;
      }
    }
  }
  outs[(size_t)q * DDIM + h * HDIM + lane] = 0.5f * accv;  // mean over bq
}

// ---------------------------------------------------------------------------
extern "C" void kernel_launch(void* const* d_in, const int* in_sizes, int n_in,
                              void* d_out, int out_size, void* d_ws, size_t ws_size,
                              hipStream_t stream) {
  const float* query = (const float*)d_in[0];   // (1, 40000, 256)
  const float* value = (const float*)d_in[1];   // (1, 80000, 256)
  const float* refp  = (const float*)d_in[2];   // (1, 40000, 1, 2)
  // d_in[3] spatial_shapes: constant [[200,200]] baked in
  const float* W_v  = (const float*)d_in[4];
  const float* b_v  = (const float*)d_in[5];
  const float* W_so = (const float*)d_in[6];
  const float* b_so = (const float*)d_in[7];
  const float* W_aw = (const float*)d_in[8];
  const float* b_aw = (const float*)d_in[9];
  const float* W_o  = (const float*)d_in[10];
  const float* b_o  = (const float*)d_in[11];
  float* out = (float*)d_out;

  char* ws = (char*)d_ws;
  size_t cur = 0;
  auto alloc = [&](size_t bytes) -> void* {
    void* p = ws + cur;
    cur = (cur + bytes + 255) & ~(size_t)255;
    return p;
  };
  float* vproj = (float*)alloc((size_t)2 * NQ * DDIM * 4);   // 82 MB
  float* soff  = (float*)alloc((size_t)NQ * 128 * 4);        // 20.5 MB
  float* awl   = (float*)alloc((size_t)NQ * 64 * 4);         // 10.2 MB
  float* outs  = (float*)alloc((size_t)NQ * DDIM * 4);       // 41 MB
  u16* Wv_hi  = (u16*)alloc((size_t)256 * 256 * 2);
  u16* Wv_lo  = (u16*)alloc((size_t)256 * 256 * 2);
  u16* Wso_hi = (u16*)alloc((size_t)512 * 128 * 2);
  u16* Wso_lo = (u16*)alloc((size_t)512 * 128 * 2);
  u16* Waw_hi = (u16*)alloc((size_t)512 * 64 * 2);
  u16* Waw_lo = (u16*)alloc((size_t)512 * 64 * 2);
  u16* Wo_hi  = (u16*)alloc((size_t)256 * 256 * 2);
  u16* Wo_lo  = (u16*)alloc((size_t)256 * 256 * 2);

  // 1) weight prep (transpose + bf16 hi/lo split)
  prep_w_kernel<<<(256 * 256 + 255) / 256, 256, 0, stream>>>(W_v,  Wv_hi,  Wv_lo,  256, 256);
  prep_w_kernel<<<(512 * 128 + 255) / 256, 256, 0, stream>>>(W_so, Wso_hi, Wso_lo, 512, 128);
  prep_w_kernel<<<(512 * 64  + 255) / 256, 256, 0, stream>>>(W_aw, Waw_hi, Waw_lo, 512, 64);
  prep_w_kernel<<<(256 * 256 + 255) / 256, 256, 0, stream>>>(W_o,  Wo_hi,  Wo_lo,  256, 256);

  // 2) vproj = value @ W_v + b_v          (80000 x 256 x 256)
  gemm_bf16x3_kernel<256, 256, false>
      <<<dim3(2 * NQ / BM, 256 / BN), 256, 0, stream>>>(
          value, value, Wv_hi, Wv_lo, b_v, nullptr, vproj, 256);

  // 3) off = [value[:nq] | query] @ W_so + b_so   (40000 x 512 x 128)
  gemm_bf16x3_kernel<512, 256, false>
      <<<dim3(NQ / BM, 128 / BN), 256, 0, stream>>>(
          value, query, Wso_hi, Wso_lo, b_so, nullptr, soff, 128);

  // 4) aw  = [value[:nq] | query] @ W_aw + b_aw   (40000 x 512 x 64)
  gemm_bf16x3_kernel<512, 256, false>
      <<<dim3(NQ / BM, 64 / BN), 256, 0, stream>>>(
          value, query, Waw_hi, Waw_lo, b_aw, nullptr, awl, 64);

  // 5) softmax + bilinear deformable sampling + bq-mean
  sample_kernel<<<NQ, 256, 0, stream>>>(vproj, soff, awl, refp, outs);

  // 6) out = outs @ W_o + b_o + query (identity)  (40000 x 256 x 256)
  gemm_bf16x3_kernel<256, 256, true>
      <<<dim3(NQ / BM, 256 / BN), 256, 0, stream>>>(
          outs, outs, Wo_hi, Wo_lo, b_o, query, out, 256);

  (void)in_sizes; (void)n_in; (void)out_size; (void)ws_size;
}
